// DinoFusion_72172630442429
// MI455X (gfx1250) — compile-verified, matrix operand order checked by reading, not run on hardware
//
#include <hip/hip_runtime.h>
#include <hip/hip_bf16.h>
#include <math.h>

// ---------------------------------------------------------------------------
// Problem constants (B=4, C=256, H=W=64, G=4, K=9)
// ---------------------------------------------------------------------------
#define CIN   256
#define HH    64
#define WW    64
#define HW    4096          // H*W
#define CG    64            // C / G
#define NOFF  108           // 27*G offset-conv out channels
#define FUSED_ELEMS  4194304u   // 4*256*4096
#define OFF_ELEMS    1179648u   // 4*72*4096

typedef __attribute__((ext_vector_type(16))) __bf16 v16bf;
typedef __attribute__((ext_vector_type(8)))  float  v8f;

// fp32 -> bf16 via hardware convert; raw bits for LDS staging
__device__ __forceinline__ unsigned short f2us(float f) {
    union { __bf16 h; unsigned short u; } cv; cv.h = (__bf16)f; return cv.u;
}
__device__ __forceinline__ float us2f(unsigned short us) {
    union { unsigned u; float f; } cv; cv.u = ((unsigned)us) << 16; return cv.f;
}

// WMMA bf16 fragment from an LDS row: elems 0..7 = row[off..off+7],
// elems 8..15 = row[off+16..off+23] (ISA 7.12.2) -> two ds_load_b128.
__device__ __forceinline__ v16bf ld_frag(const unsigned short* row, int off) {
    union { v16bf v; uint4 q[2]; } u;
    u.q[0] = *(const uint4*)(row + off);
    u.q[1] = *(const uint4*)(row + off + 16);
    return u.v;
}

// ---------------------------------------------------------------------------
// Kernel 0: one-time weight repack to bf16 (removes per-block re-conversion).
//   qkvpack [which][m][c]           : 3*256*256
//   wpack   [tap][m(112 pad)][ic]   : 9*112*512  (zero-padded m>=108)
//   dpack   [g*64+m][tap][i]        : 256*9*64
// ---------------------------------------------------------------------------
#define WPACK_N  (9 * 112 * 512)    // 516096
#define DPACK_N  (256 * 576)        // 147456
#define QPACK_N  (3 * 256 * 256)    // 196608
#define PACK_TOT (WPACK_N + DPACK_N + QPACK_N)   // 860160 = 3360*256

__global__ void __launch_bounds__(256)
pack_weights(const float* __restrict__ wq, const float* __restrict__ wk,
             const float* __restrict__ wv, const float* __restrict__ w_off,
             const float* __restrict__ dcn_w,
             unsigned short* __restrict__ wpack,
             unsigned short* __restrict__ dpack,
             unsigned short* __restrict__ qkvpack) {
    const int idx = blockIdx.x * 256 + threadIdx.x;
    if (idx < WPACK_N) {
        const int tap = idx / (112 * 512);
        const int r   = idx % (112 * 512);
        const int m   = r >> 9;
        const int ic  = r & 511;
        const int ky = tap / 3, kx = tap % 3;
        wpack[idx] = (m < NOFF)
            ? f2us(w_off[(((size_t)m * 512 + ic) * 3 + ky) * 3 + kx])
            : (unsigned short)0;
    } else if (idx < WPACK_N + DPACK_N) {
        const int j = idx - WPACK_N;
        const int m = j / 576;          // 0..255 (g*64 + mg)
        const int r = j % 576;
        const int k = r >> 6;
        const int i = r & 63;
        dpack[j] = f2us(dcn_w[((size_t)m * CG + i) * 9 + k]);
    } else if (idx < PACK_TOT) {
        const int j     = idx - WPACK_N - DPACK_N;
        const int which = j >> 16;      // /65536
        const int mc    = j & 65535;
        const float* w = (which == 0) ? wq : (which == 1) ? wk : wv;
        qkvpack[j] = f2us(w[mc]);
    }
}

// ---------------------------------------------------------------------------
// Kernel 1: Q/K/V 1x1 convs. Block = 512 thr = 16 waves (one M tile each),
// N tile = 64 pixels. Per 32-deep K chunk: A copied bf16->LDS via uint4,
// B (fp32 activations) loaded float4 + packed-cvt to LDS; then each wave
// does 4 WMMAs (A fragment reused across 4 pixel subtiles).
// ---------------------------------------------------------------------------
__global__ void __launch_bounds__(512)
qkv_conv1x1_wmma(const float* __restrict__ rife,
                 const float* __restrict__ d0,
                 const float* __restrict__ d1,
                 const float* __restrict__ bq,
                 const float* __restrict__ bk,
                 const float* __restrict__ bv,
                 const unsigned short* __restrict__ qkvpack,
                 unsigned short* __restrict__ comb,
                 unsigned short* __restrict__ vbuf) {
    __shared__ __align__(16) unsigned short As[256 * 32];  // [m][k] 16KB
    __shared__ __align__(16) unsigned short Bs[64 * 32];   // [p][k] 4KB

    const int tid   = threadIdx.x;
    const int lane  = tid & 31;
    const int wave  = tid >> 5;                  // M tile 0..15
    const int ptile = blockIdx.x;                // 0..511 (8*HW/64)
    const int which = blockIdx.z;                // 0=Q 1=K 2=V

    const float* bias = (which == 0) ? bq : (which == 1) ? bk : bv;
    const unsigned short* wp = qkvpack + (size_t)which * 256 * 256;

    const int gp0   = ptile * 64;                // tile never crosses a batch
    const int bb    = gp0 >> 12;
    const int pbase = gp0 & 4095;
    const float* x;
    if (which == 0) x = rife + (size_t)(bb & 3) * CIN * HW;
    else            x = (bb < 4) ? d0 + (size_t)bb * CIN * HW
                                 : d1 + (size_t)(bb - 4) * CIN * HW;

    const int kOff = (lane < 16) ? 0 : 8;
    const int n    = lane & 15;

    v8f acc[4] = {};
    for (int kb = 0; kb < CIN; kb += 32) {
        // ---- stage A: 256x32 bf16, pure uint4 copies (2 per thread)
        {
            uint4* dstA = (uint4*)As;
#pragma unroll
            for (int t2 = 0; t2 < 2; ++t2) {
                const int q  = tid * 2 + t2;     // 0..1023
                const int m  = q >> 2;
                const int qi = q & 3;
                dstA[q] = *(const uint4*)(wp + (size_t)m * CIN + kb + qi * 8);
            }
        }
        // ---- stage B: 32c x 64p, one float4 (along p) per thread
        {
            const int c  = tid >> 4;             // 0..31
            const int p4 = (tid & 15) * 4;       // 0..60
            const float4 f = *(const float4*)(x + (size_t)(kb + c) * HW + pbase + p4);
            Bs[(p4 + 0) * 32 + c] = f2us(f.x);
            Bs[(p4 + 1) * 32 + c] = f2us(f.y);
            Bs[(p4 + 2) * 32 + c] = f2us(f.z);
            Bs[(p4 + 3) * 32 + c] = f2us(f.w);
        }
        __syncthreads();

        const v16bf a = ld_frag(As + (wave * 16 + n) * 32, kOff);
#pragma unroll
        for (int s = 0; s < 4; ++s) {
            const v16bf b = ld_frag(Bs + (s * 16 + n) * 32, kOff);
            acc[s] = __builtin_amdgcn_wmma_f32_16x16x32_bf16(false, a, false, b,
                                                             (short)0, acc[s], false, false);
        }
        __syncthreads();
    }

    const int mOff = (lane < 16) ? 0 : 8;
#pragma unroll
    for (int s = 0; s < 4; ++s) {
        const int col = pbase + s * 16 + n;
#pragma unroll
        for (int r = 0; r < 8; ++r) {
            const int oc  = wave * 16 + mOff + r;
            const float v = acc[s][r] + bias[oc];
            if (which < 2)
                comb[((size_t)bb * 512 + which * 256 + oc) * HW + col] = f2us(v);
            else
                vbuf[((size_t)bb * 256 + oc) * HW + col] = f2us(v);
        }
    }
}

// ---------------------------------------------------------------------------
// Kernel 2: 3x3 offset conv, implicit GEMM (M=108 pad 112, K=512*9).
// Block = 224 thr = 7 waves; N tile = 32 pixels (2 WMMAs per chunk).
// A staged via pure uint4 copies from pre-packed weights.
// Epilogue: ch 0..71 -> offsets into d_out, ch 72..107 -> sigmoid mask.
// ---------------------------------------------------------------------------
__global__ void __launch_bounds__(224)
off_conv3x3_wmma(const unsigned short* __restrict__ comb,
                 const unsigned short* __restrict__ wpack,
                 const float* __restrict__ b_off,
                 float* __restrict__ dout,
                 float* __restrict__ maskbuf) {
    __shared__ __align__(16) unsigned short As[112 * 32];  // [m][k] 7KB
    __shared__ __align__(16) unsigned short Bs[32 * 32];   // [p][k] 2KB

    const int tid   = threadIdx.x;
    const int lane  = tid & 31;
    const int wave  = tid >> 5;                  // M tile 0..6
    const int ptile = blockIdx.x;                // 0..1023 (8*HW/32)

    const int gp0   = ptile * 32;
    const int bb    = gp0 >> 12;
    const int pbase = gp0 & 4095;
    const int y     = pbase >> 6;
    const int xpix  = pbase & 63;                // 0 or 32; tile spans one row
    const unsigned short* cb = comb + (size_t)bb * 512 * HW;

    const int kOff = (lane < 16) ? 0 : 8;
    const int n    = lane & 15;

    v8f acc[2] = {};
    for (int ky = 0; ky < 3; ++ky) {
        const int yy   = y + ky - 1;
        const bool yok = (unsigned)yy < (unsigned)HH;
        for (int kx = 0; kx < 3; ++kx) {
            const unsigned short* wtap = wpack + (size_t)(ky * 3 + kx) * 112 * 512;
            for (int icb = 0; icb < 512; icb += 32) {
                // ---- stage A: 112x32 bf16, pure uint4 copies (2 per thread)
                {
                    uint4* dstA = (uint4*)As;
#pragma unroll
                    for (int t2 = 0; t2 < 2; ++t2) {
                        const int q  = tid * 2 + t2;   // 0..447
                        const int m  = q >> 2;
                        const int qi = q & 3;
                        dstA[q] = *(const uint4*)(wtap + (size_t)m * 512 + icb + qi * 8);
                    }
                }
                // ---- stage B: 32 ic x 32 p with shift + zero padding
                for (int idx = tid; idx < 1024; idx += 224) {
                    const int c  = idx >> 5;      // 0..31
                    const int p  = idx & 31;
                    const int xx = xpix + p + kx - 1;
                    const bool ok = yok && ((unsigned)xx < (unsigned)WW);
                    Bs[p * 32 + c] = ok ? cb[(size_t)(icb + c) * HW + yy * WW + xx]
                                        : (unsigned short)0;
                }
                __syncthreads();

                const v16bf a = ld_frag(As + (wave * 16 + n) * 32, kOff);
#pragma unroll
                for (int s = 0; s < 2; ++s) {
                    const v16bf b = ld_frag(Bs + (s * 16 + n) * 32, kOff);
                    acc[s] = __builtin_amdgcn_wmma_f32_16x16x32_bf16(
                        false, a, false, b, (short)0, acc[s], false, false);
                }
                __syncthreads();
            }
        }
    }

    const int mOff = (lane < 16) ? 0 : 8;
#pragma unroll
    for (int s = 0; s < 2; ++s) {
        const int col = pbase + s * 16 + n;
#pragma unroll
        for (int r = 0; r < 8; ++r) {
            const int oc = wave * 16 + mOff + r;
            if (oc < NOFF) {
                const float v = acc[s][r] + b_off[oc];
                if (oc < 72) {   // offset channels (reference outputs 2 and 3)
                    const size_t base = (size_t)FUSED_ELEMS +
                        ((bb < 4) ? ((size_t)bb * 72 + oc) * HW
                                  : (size_t)OFF_ELEMS + ((size_t)(bb - 4) * 72 + oc) * HW);
                    dout[base + col] = v;
                } else {         // mask logits -> sigmoid
                    const float sg = 1.0f / (1.0f + __expf(-v));
                    maskbuf[((size_t)bb * 36 + (oc - 72)) * HW + col] = sg;
                }
            }
        }
    }
}

// ---------------------------------------------------------------------------
// Kernel 3: deformable conv. Pre-packed group weights copied to LDS (72KB)
// with uint4 moves; bilinear gather of bf16 V builds B; per-group GEMM
// M=64, K=9*64 via WMMA. Block covers 4 pixel tiles.
// ---------------------------------------------------------------------------
__global__ void __launch_bounds__(128)
deform_conv_wmma(const unsigned short* __restrict__ vbuf,
                 const unsigned short* __restrict__ dpack,
                 const float* __restrict__ maskbuf,
                 const float* __restrict__ dout,   // offsets live here
                 float* __restrict__ aligned) {
    __shared__ __align__(16) unsigned short Wsm[64 * 9 * 64];  // [m][k][i] 72KB

    const int tid  = threadIdx.x;
    const int lane = tid & 31;
    const int wave = tid >> 5;                   // M tile within group 0..3
    const int bg   = blockIdx.y;                 // 0..31
    const int bb   = bg >> 2;
    const int g    = bg & 3;

    // ---- stage group weights once: pure uint4 copies (36 per thread)
    {
        uint4* dst = (uint4*)Wsm;
        const uint4* src = (const uint4*)(dpack + (size_t)g * 64 * 576);
        for (int idx = tid; idx < (64 * 576) / 8; idx += 128) dst[idx] = src[idx];
    }
    __syncthreads();

    const unsigned short* Vg = vbuf + ((size_t)bb * 256 + g * CG) * HW;
    const size_t offBase = (size_t)FUSED_ELEMS +
        ((bb < 4) ? (size_t)bb * 72 * HW
                  : (size_t)OFF_ELEMS + (size_t)(bb - 4) * 72 * HW);

    const int kOff = (lane < 16) ? 0 : 8;
    const int n    = lane & 15;
    const unsigned short* arow = Wsm + (size_t)(wave * 16 + n) * 9 * 64;

    for (int pt = 0; pt < 4; ++pt) {
        const int ptile = blockIdx.x * 4 + pt;   // 0..255 within batch
        const int p = ptile * 16 + n;
        const int y = p >> 6;
        const int x = p & 63;

        v8f acc = {};
        for (int k = 0; k < 9; ++k) {
            const int ky = k / 3, kx = k % 3;
            const float offy = dout[offBase + (size_t)((g * 9 + k) * 2 + 0) * HW + p];
            const float offx = dout[offBase + (size_t)((g * 9 + k) * 2 + 1) * HW + p];
            const float m    = maskbuf[((size_t)bb * 36 + g * 9 + k) * HW + p];

            const float py = offy + (float)(y - 1 + ky);
            const float px = offx + (float)(x - 1 + kx);
            const float y0f = floorf(py), x0f = floorf(px);
            const float fy = py - y0f, fx = px - x0f;
            const int y0 = (int)y0f, x0 = (int)x0f;

            const bool v00 = ((unsigned)y0       < (unsigned)HH) && ((unsigned)x0       < (unsigned)WW);
            const bool v01 = ((unsigned)y0       < (unsigned)HH) && ((unsigned)(x0 + 1) < (unsigned)WW);
            const bool v10 = ((unsigned)(y0 + 1) < (unsigned)HH) && ((unsigned)x0       < (unsigned)WW);
            const bool v11 = ((unsigned)(y0 + 1) < (unsigned)HH) && ((unsigned)(x0 + 1) < (unsigned)WW);

            const float w00 = (1.0f - fy) * (1.0f - fx) * m * (v00 ? 1.0f : 0.0f);
            const float w01 = (1.0f - fy) * fx          * m * (v01 ? 1.0f : 0.0f);
            const float w10 = fy * (1.0f - fx)          * m * (v10 ? 1.0f : 0.0f);
            const float w11 = fy * fx                   * m * (v11 ? 1.0f : 0.0f);

            const int yc0 = min(max(y0, 0), HH - 1),     xc0 = min(max(x0, 0), WW - 1);
            const int yc1 = min(max(y0 + 1, 0), HH - 1), xc1 = min(max(x0 + 1, 0), WW - 1);
            const int i00 = yc0 * WW + xc0, i01 = yc0 * WW + xc1;
            const int i10 = yc1 * WW + xc0, i11 = yc1 * WW + xc1;

#pragma unroll
            for (int ih = 0; ih < 2; ++ih) {
                const v16bf a = ld_frag(arow + k * 64, ih * 32 + kOff);
                v16bf b;
#pragma unroll
                for (int e = 0; e < 16; ++e) {
                    const int i = ih * 32 + kOff + (e < 8 ? e : e + 8);
                    const unsigned short* ch = Vg + (size_t)i * HW;
                    const float s = w00 * us2f(ch[i00]) + w01 * us2f(ch[i01]) +
                                    w10 * us2f(ch[i10]) + w11 * us2f(ch[i11]);
                    b[e] = (__bf16)s;
                }
                acc = __builtin_amdgcn_wmma_f32_16x16x32_bf16(false, a, false, b,
                                                              (short)0, acc, false, false);
            }
        }

        const int mOff = (lane < 16) ? 0 : 8;
#pragma unroll
        for (int r = 0; r < 8; ++r) {
            const int oc = g * CG + wave * 16 + mOff + r;
            aligned[((size_t)bb * 256 + oc) * HW + (ptile * 16 + n)] = acc[r];
        }
    }
}

// ---------------------------------------------------------------------------
// Kernel 4: fused = rife + gamma * (aligned[0:4] + aligned[4:8]), float4-wide.
// ---------------------------------------------------------------------------
__global__ void __launch_bounds__(256)
residual_fuse(const float* __restrict__ rife,
              const float* __restrict__ aligned,
              const float* __restrict__ gamma,
              float* __restrict__ dout) {
    const unsigned idx = (blockIdx.x * 256u + threadIdx.x) * 4u;
    if (idx < FUSED_ELEMS) {
        const float4 r  = *(const float4*)(rife + idx);
        const float4 a0 = *(const float4*)(aligned + idx);
        const float4 a1 = *(const float4*)(aligned + idx + FUSED_ELEMS);
        const float gm  = gamma[0];
        float4 o;
        o.x = r.x + gm * (a0.x + a1.x);
        o.y = r.y + gm * (a0.y + a1.y);
        o.z = r.z + gm * (a0.z + a1.z);
        o.w = r.w + gm * (a0.w + a1.w);
        *(float4*)(dout + idx) = o;
    }
}

// ---------------------------------------------------------------------------
// Launch
// ---------------------------------------------------------------------------
extern "C" void kernel_launch(void* const* d_in, const int* in_sizes, int n_in,
                              void* d_out, int out_size, void* d_ws, size_t ws_size,
                              hipStream_t stream) {
    (void)in_sizes; (void)n_in; (void)out_size; (void)ws_size;

    const float* rife  = (const float*)d_in[0];
    const float* d0    = (const float*)d_in[1];
    const float* d1    = (const float*)d_in[2];
    const float* wq    = (const float*)d_in[3];
    const float* bq    = (const float*)d_in[4];
    const float* wk    = (const float*)d_in[5];
    const float* bk    = (const float*)d_in[6];
    const float* wv    = (const float*)d_in[7];
    const float* bv    = (const float*)d_in[8];
    const float* w_off = (const float*)d_in[9];
    const float* b_off = (const float*)d_in[10];
    const float* dcn_w = (const float*)d_in[11];
    const float* gamma = (const float*)d_in[12];
    float* dout = (float*)d_out;

    // Workspace layout (bytes):
    //   comb    bf16: 33,554,432   @ 0
    //   vbuf    bf16: 16,777,216   @ 33,554,432
    //   maskbuf f32 :  4,718,592   @ 50,331,648
    //   aligned f32 : 33,554,432   @ 55,050,240
    //   wpack   bf16:  1,032,192   @ 88,604,672
    //   dpack   bf16:    294,912   @ 89,636,864
    //   qkvpack bf16:    393,216   @ 89,931,776
    char* ws = (char*)d_ws;
    unsigned short* comb    = (unsigned short*)(ws + 0);
    unsigned short* vbuf    = (unsigned short*)(ws + 33554432u);
    float*          maskbuf = (float*)(ws + 50331648u);
    float*          aligned = (float*)(ws + 55050240u);
    unsigned short* wpack   = (unsigned short*)(ws + 88604672u);
    unsigned short* dpack   = (unsigned short*)(ws + 89636864u);
    unsigned short* qkvpack = (unsigned short*)(ws + 89931776u);

    // K0: one-time bf16 weight repack.
    pack_weights<<<dim3(PACK_TOT / 256u), 256, 0, stream>>>(
        wq, wk, wv, w_off, dcn_w, wpack, dpack, qkvpack);

    // K1: Q/K/V 1x1 convs. 512 pixel tiles (64 px each) x {Q,K,V}.
    qkv_conv1x1_wmma<<<dim3(512, 1, 3), 512, 0, stream>>>(
        rife, d0, d1, bq, bk, bv, qkvpack, comb, vbuf);

    // K2: 3x3 offset conv; 1024 pixel tiles (32 px each).
    off_conv3x3_wmma<<<dim3(1024, 1, 1), 224, 0, stream>>>(
        comb, wpack, b_off, dout, maskbuf);

    // K3: deformable conv per (batch-slice, group); 4 pixel tiles per block.
    deform_conv_wmma<<<dim3(64, 32, 1), 128, 0, stream>>>(
        vbuf, dpack, maskbuf, dout, aligned);

    // K4: residual fuse into d_out[0 .. FUSED_ELEMS).
    residual_fuse<<<dim3(FUSED_ELEMS / 1024u), 256, 0, stream>>>(
        rife, aligned, gamma, dout);
}
